// AMOE_79843442033161
// MI455X (gfx1250) — compile-verified, matrix operand order checked by reading, not run on hardware
//
#include <hip/hip_runtime.h>
#include <hip/hip_bf16.h>

#define DEV __device__ __forceinline__

typedef __attribute__((ext_vector_type(16))) __bf16    v16bf;
typedef __attribute__((ext_vector_type(8)))  float     v8f;
typedef __attribute__((ext_vector_type(4)))  unsigned  v4u;
typedef __attribute__((ext_vector_type(8)))  int       v8i;
typedef __attribute__((ext_vector_type(4)))  int       v4i;

#if defined(__has_builtin)
#if __has_builtin(__builtin_amdgcn_tensor_load_to_lds) && __has_builtin(__builtin_amdgcn_s_wait_tensorcnt)
#define USE_TDM 1
#endif
#endif
#ifndef USE_TDM
#define USE_TDM 0
#endif

namespace {
constexpr int Bn = 8, NPATCH = 256, PDIM = 768;
constexpr int D = 1024, H = 16, HD = 64;
constexpr int L = 4, E = 8, F = 1024, NS = 4;
constexpr int S  = 1 + NS + NPATCH;   // 261 tokens per image
constexpr int Sp = 288;               // padded seq len (multiple of 32)
constexpr int Tp = Bn * Sp;           // 2304 padded rows
constexpr float EPSF  = 1e-6f;
constexpr float THETA = 10000.0f;

constexpr int BM = 256;               // block macro-tile rows (8 waves x 32)
constexpr int BN = 64;                // block macro-tile cols
constexpr int LDSB = 65;              // padded LDS row stride in floats (bank-conflict-free)
}

// ---------- bf16 helpers ----------
DEV __bf16 f2bf(float f) { return (__bf16)f; }   // native cvt on gfx1250

DEV v8f vzero8() { v8f z = {0.f,0.f,0.f,0.f,0.f,0.f,0.f,0.f}; return z; }

DEV v8f bfwmma(v16bf a, v16bf b, v8f c) {
  return __builtin_amdgcn_wmma_f32_16x16x32_bf16(
      false, a, false, b, (short)0, c, false, false);
}

// A-fragment: 16x32 bf16 tile from row-major source (also K^T score B-frag).
// Lane L holds row M=L&15; K split {0..7|8..15},{16..23|24..31} by lane half.
DEV v16bf load_a_frag(const float* __restrict__ Ap, int lda, int k0, int lane, float scale) {
  int r    = lane & 15;
  int half = (lane >> 4) & 1;
  const float* p0 = Ap + (size_t)r * lda + k0 + 8 * half;
  const float* p1 = p0 + 16;
  v16bf a;
#pragma unroll
  for (int i = 0; i < 8; ++i) {
    a[i]     = f2bf(p0[i] * scale);
    a[i + 8] = f2bf(p1[i] * scale);
  }
  return a;
}

// B-fragment straight from row-major global [K,N] (used by attention P@V).
DEV v16bf load_b_strided(const float* __restrict__ Bp, int ldb, int col, int lane) {
  int n    = col + (lane & 15);
  int half = (lane >> 4) & 1;
  v16bf v;
#pragma unroll
  for (int e = 0; e < 16; ++e) {
    int kk = (e & 7) + 8 * (((e >> 3) << 1) + half);
    v[e] = f2bf(Bp[(size_t)kk * ldb + n]);
  }
  return v;
}

// B-fragment from an LDS-staged 32xBN tile with padded row stride LDSB.
DEV v16bf lds_b_frag(const float* Lb, int col, int lane) {
  int n    = col + (lane & 15);
  int half = (lane >> 4) & 1;
  v16bf v;
#pragma unroll
  for (int e = 0; e < 16; ++e) {
    int kk = (e & 7) + 8 * (((e >> 3) << 1) + half);
    v[e] = f2bf(Lb[kk * LDSB + n]);
  }
  return v;
}

// ---------- Tensor Data Mover: DMA a 32xBN f32 tile into LDS ----------
// Uses D# pad feature (pad_interval=64 DWORDs, pad_amount=1 DWORD) so the tile
// lands in LDS with a 65-float row stride -> conflict-free B-fragment reads.
#if USE_TDM
DEV void tdm_load_tile(unsigned lds_byte_addr, const float* gsrc, int ld_elems) {
  unsigned long long ga = (unsigned long long)gsrc;
  v4u g0;
  g0[0] = 1u;                                            // count=1 (valid user D#)
  g0[1] = lds_byte_addr;                                 // lds_addr
  g0[2] = (unsigned)(ga & 0xFFFFFFFFu);                  // global_addr[31:0]
  g0[3] = (unsigned)((ga >> 32) & 0x01FFFFFFu) | (2u << 30); // addr[56:32] | type=2
  v8i g1;
  // data_size=4B (2), pad_enable=1, pad_interval=64 DWORDs (5), pad_amount=1 (0)
  g1[0] = (int)((2u << 16) | (1u << 20) | (5u << 22));
  g1[1] = (int)(((unsigned)ld_elems & 0xFFFFu) << 16);   // tensor_dim0[15:0]
  g1[2] = (int)((((unsigned)ld_elems >> 16) & 0xFFFFu) | (0x8000u << 16)); // dim0 hi | tensor_dim1 lo (big)
  g1[3] = (int)(((unsigned)BN) << 16);                   // tensor_dim1 hi=0 | tile_dim0=BN
  g1[4] = 32;                                            // tile_dim1=32, tile_dim2=0
  g1[5] = ld_elems;                                      // tensor_dim0_stride[31:0]
  g1[6] = 0;                                             // stride hi | dim1_stride lo
  g1[7] = 0;
  v4i z4 = {0, 0, 0, 0};
  v8i z8 = {0, 0, 0, 0, 0, 0, 0, 0};
  // clang-23 / therock signature: (v4u, v8i, v4i, v4i, v8i, i32 cpol)
  __builtin_amdgcn_tensor_load_to_lds(g0, g1, z4, z4, z8, 0);
}
#endif

// Cooperative fallback staging (all 256 threads), same padded layout.
DEV void stage_b_tile(float* dstLds, const float* __restrict__ gsrc, int ld_elems, int tid) {
  for (int i = tid; i < 32 * BN; i += 256) {
    int r = i >> 6, c = i & (BN - 1);
    dstLds[r * LDSB + c] = gsrc[(size_t)r * ld_elems + c];
  }
}

// ---------- trivial utility kernels ----------
__global__ __launch_bounds__(256) void k_zero(float* __restrict__ p, size_t n) {
  for (size_t i = blockIdx.x * (size_t)blockDim.x + threadIdx.x; i < n;
       i += (size_t)gridDim.x * blockDim.x) p[i] = 0.f;
}

__global__ __launch_bounds__(256) void k_copy(const float* __restrict__ src,
                                              float* __restrict__ dst, size_t n) {
  for (size_t i = blockIdx.x * (size_t)blockDim.x + threadIdx.x; i < n;
       i += (size_t)gridDim.x * blockDim.x) dst[i] = src[i];
}

__global__ __launch_bounds__(256) void k_fill(float* __restrict__ X,
                                              const float* __restrict__ cls,
                                              const float* __restrict__ st) {
  int slot = blockIdx.x % (1 + NS);
  int b    = blockIdx.x / (1 + NS);
  const float* src = (slot == 0) ? cls : st + (size_t)(slot - 1) * D;
  float* dst = X + (size_t)(b * Sp + slot) * D;
  for (int i = threadIdx.x; i < D; i += blockDim.x) dst[i] = src[i];
}

// ---------- RMSNorm ----------
__global__ __launch_bounds__(256) void k_rmsnorm(const float* __restrict__ X,
                                                 const float* __restrict__ W,
                                                 float* __restrict__ Y, int nd) {
  int row = blockIdx.x;
  const float* x = X + (size_t)row * nd;
  float ss = 0.f;
  for (int i = threadIdx.x; i < nd; i += blockDim.x) { float v = x[i]; ss += v * v; }
#pragma unroll
  for (int m = 1; m <= 16; m <<= 1) ss += __shfl_xor(ss, m, 32);
  __shared__ float red[8];
  int wid = threadIdx.x >> 5, lane = threadIdx.x & 31;
  if (lane == 0) red[wid] = ss;
  __syncthreads();
  if (threadIdx.x == 0) {
    float t = 0.f;
#pragma unroll
    for (int w = 0; w < 8; ++w) t += red[w];
    red[0] = rsqrtf(t / nd + EPSF);
  }
  __syncthreads();
  float inv = red[0];
  for (int i = threadIdx.x; i < nd; i += blockDim.x)
    Y[(size_t)row * nd + i] = x[i] * W[i] * inv;
}

// ---------- final RMSNorm -> packed output ----------
__global__ __launch_bounds__(256) void k_final(const float* __restrict__ X,
                                               const float* __restrict__ W,
                                               float* __restrict__ out) {
  int t = blockIdx.x;
  int b = t / S, s = t % S;
  const float* x = X + (size_t)(b * Sp + s) * D;
  float ss = 0.f;
  for (int i = threadIdx.x; i < D; i += blockDim.x) { float v = x[i]; ss += v * v; }
#pragma unroll
  for (int m = 1; m <= 16; m <<= 1) ss += __shfl_xor(ss, m, 32);
  __shared__ float red[8];
  int wid = threadIdx.x >> 5, lane = threadIdx.x & 31;
  if (lane == 0) red[wid] = ss;
  __syncthreads();
  if (threadIdx.x == 0) {
    float tt = 0.f;
#pragma unroll
    for (int w = 0; w < 8; ++w) tt += red[w];
    red[0] = rsqrtf(tt / D + EPSF);
  }
  __syncthreads();
  float inv = red[0];
  for (int i = threadIdx.x; i < D; i += blockDim.x)
    out[(size_t)t * D + i] = x[i] * W[i] * inv;
}

// ---------- RoPE (in place on q and k) ----------
__global__ __launch_bounds__(256) void k_rope(float* __restrict__ Q, float* __restrict__ Kq) {
  int row = blockIdx.x;
  int s   = row % Sp;
  for (int p = threadIdx.x; p < H * (HD / 2); p += blockDim.x) {
    int hh = p / (HD / 2), i = p % (HD / 2);
    float freq = powf(THETA, -(2.0f * i) / (float)HD);
    float ang  = (float)s * freq;
    float sn, c;
    sincosf(ang, &sn, &c);
    size_t base = (size_t)row * D + hh * HD + 2 * i;
    float qr = Q[base], qi = Q[base + 1];
    Q[base]     = qr * c - qi * sn;
    Q[base + 1] = qr * sn + qi * c;
    float kr = Kq[base], ki = Kq[base + 1];
    Kq[base]     = kr * c - ki * sn;
    Kq[base + 1] = kr * sn + ki * c;
  }
}

// =====================================================================
// GEMM: block computes BM x BN; 8 waves stacked in M share one LDS-staged
// B tile (32 x BN per k-step), TDM double-buffered. Each wave: 32x64 tile,
// 2 A-frags x 4 B-frags -> 8 WMMAs per k-step.
// =====================================================================
__global__ __launch_bounds__(256) void k_gemm(const float* __restrict__ A, int lda,
                                              const float* __restrict__ Bw, int ldb,
                                              float* __restrict__ C, int ldc,
                                              int K, const float* __restrict__ res) {
  const int wave = threadIdx.x >> 5, lane = threadIdx.x & 31;
  const int tn0 = blockIdx.x * BN;
  const int tm0 = blockIdx.y * BM + wave * 32;
  __shared__ float Bt[2][32 * LDSB];

  v8f acc[2][4];
#pragma unroll
  for (int i = 0; i < 2; ++i)
#pragma unroll
    for (int j = 0; j < 4; ++j) acc[i][j] = vzero8();

#if USE_TDM
  if (threadIdx.x < 32)
    tdm_load_tile((unsigned)(unsigned long long)&Bt[0][0], Bw + tn0, ldb);
#endif

  const float* Arow0 = A + (size_t)tm0 * lda;
  const float* Arow1 = A + (size_t)(tm0 + 16) * lda;

  for (int k0 = 0; k0 < K; k0 += 32) {
    const int cur = (k0 >> 5) & 1;
#if USE_TDM
    if (threadIdx.x < 32) {
      if (k0 + 32 < K) {
        tdm_load_tile((unsigned)(unsigned long long)&Bt[cur ^ 1][0],
                      Bw + (size_t)(k0 + 32) * ldb + tn0, ldb);
        __builtin_amdgcn_s_wait_tensorcnt(1);
      } else {
        __builtin_amdgcn_s_wait_tensorcnt(0);
      }
    }
    __syncthreads();
#else
    stage_b_tile(&Bt[cur][0], Bw + (size_t)k0 * ldb + tn0, ldb, threadIdx.x);
    __syncthreads();
#endif
    if (k0 + 32 < K) __builtin_prefetch(Arow0 + k0 + 32, 0, 1);
    v16bf a0 = load_a_frag(Arow0, lda, k0, lane, 1.0f);
    v16bf a1 = load_a_frag(Arow1, lda, k0, lane, 1.0f);
#pragma unroll
    for (int j = 0; j < 4; ++j) {
      v16bf bf = lds_b_frag(&Bt[cur][0], j * 16, lane);
      acc[0][j] = bfwmma(a0, bf, acc[0][j]);
      acc[1][j] = bfwmma(a1, bf, acc[1][j]);
    }
    __syncthreads();
  }

  const int n_base = tn0 + (lane & 15);
  const int m_base = tm0 + ((lane >> 4) << 3);
#pragma unroll
  for (int i = 0; i < 2; ++i)
#pragma unroll
    for (int j = 0; j < 4; ++j)
#pragma unroll
      for (int r = 0; r < 8; ++r) {
        size_t idx = (size_t)(m_base + 16 * i + r) * ldc + n_base + 16 * j;
        C[idx] = (res ? res[idx] : 0.f) + acc[i][j][r];
      }
}

// ---------- MoE up: HH = silu(A@W1) * (A@W3), two TDM streams ----------
__global__ __launch_bounds__(256) void k_moe_up(const float* __restrict__ A,
                                                const float* __restrict__ W1,
                                                const float* __restrict__ W3,
                                                float* __restrict__ HH, int K) {
  const int wave = threadIdx.x >> 5, lane = threadIdx.x & 31;
  const int tn0 = blockIdx.x * BN;
  const int tm0 = blockIdx.y * BM + wave * 32;
  __shared__ float B1[2][32 * LDSB];
  __shared__ float B3[2][32 * LDSB];

  v8f ac1[2][4], ac3[2][4];
#pragma unroll
  for (int i = 0; i < 2; ++i)
#pragma unroll
    for (int j = 0; j < 4; ++j) { ac1[i][j] = vzero8(); ac3[i][j] = vzero8(); }

#if USE_TDM
  if (threadIdx.x < 32) {
    tdm_load_tile((unsigned)(unsigned long long)&B1[0][0], W1 + tn0, F);
    tdm_load_tile((unsigned)(unsigned long long)&B3[0][0], W3 + tn0, F);
  }
#endif
  const float* Arow0 = A + (size_t)tm0 * K;
  const float* Arow1 = A + (size_t)(tm0 + 16) * K;

  for (int k0 = 0; k0 < K; k0 += 32) {
    const int cur = (k0 >> 5) & 1;
#if USE_TDM
    if (threadIdx.x < 32) {
      if (k0 + 32 < K) {
        tdm_load_tile((unsigned)(unsigned long long)&B1[cur ^ 1][0],
                      W1 + (size_t)(k0 + 32) * F + tn0, F);
        tdm_load_tile((unsigned)(unsigned long long)&B3[cur ^ 1][0],
                      W3 + (size_t)(k0 + 32) * F + tn0, F);
        __builtin_amdgcn_s_wait_tensorcnt(2);
      } else {
        __builtin_amdgcn_s_wait_tensorcnt(0);
      }
    }
    __syncthreads();
#else
    stage_b_tile(&B1[cur][0], W1 + (size_t)k0 * F + tn0, F, threadIdx.x);
    stage_b_tile(&B3[cur][0], W3 + (size_t)k0 * F + tn0, F, threadIdx.x);
    __syncthreads();
#endif
    v16bf a0 = load_a_frag(Arow0, K, k0, lane, 1.0f);
    v16bf a1 = load_a_frag(Arow1, K, k0, lane, 1.0f);
#pragma unroll
    for (int j = 0; j < 4; ++j) {
      v16bf b1 = lds_b_frag(&B1[cur][0], j * 16, lane);
      ac1[0][j] = bfwmma(a0, b1, ac1[0][j]);
      ac1[1][j] = bfwmma(a1, b1, ac1[1][j]);
      v16bf b3 = lds_b_frag(&B3[cur][0], j * 16, lane);
      ac3[0][j] = bfwmma(a0, b3, ac3[0][j]);
      ac3[1][j] = bfwmma(a1, b3, ac3[1][j]);
    }
    __syncthreads();
  }

  const int n_base = tn0 + (lane & 15);
  const int m_base = tm0 + ((lane >> 4) << 3);
#pragma unroll
  for (int i = 0; i < 2; ++i)
#pragma unroll
    for (int j = 0; j < 4; ++j)
#pragma unroll
      for (int r = 0; r < 8; ++r) {
        float h1   = ac1[i][j][r];
        float silu = h1 / (1.f + expf(-h1));
        HH[(size_t)(m_base + 16 * i + r) * F + n_base + 16 * j] = silu * ac3[i][j][r];
      }
}

// ---------- MoE down: X += (HH@W2) * gate[row] ----------
__global__ __launch_bounds__(256) void k_moe_down(const float* __restrict__ HH,
                                                  const float* __restrict__ W2,
                                                  float* __restrict__ X,
                                                  const float* __restrict__ grow, // gate+e, stride E
                                                  int K) {
  const int wave = threadIdx.x >> 5, lane = threadIdx.x & 31;
  const int tn0 = blockIdx.x * BN;
  const int tm0 = blockIdx.y * BM + wave * 32;
  __shared__ float Bt[2][32 * LDSB];

  v8f acc[2][4];
#pragma unroll
  for (int i = 0; i < 2; ++i)
#pragma unroll
    for (int j = 0; j < 4; ++j) acc[i][j] = vzero8();

#if USE_TDM
  if (threadIdx.x < 32)
    tdm_load_tile((unsigned)(unsigned long long)&Bt[0][0], W2 + tn0, D);
#endif
  const float* Arow0 = HH + (size_t)tm0 * K;
  const float* Arow1 = HH + (size_t)(tm0 + 16) * K;

  for (int k0 = 0; k0 < K; k0 += 32) {
    const int cur = (k0 >> 5) & 1;
#if USE_TDM
    if (threadIdx.x < 32) {
      if (k0 + 32 < K) {
        tdm_load_tile((unsigned)(unsigned long long)&Bt[cur ^ 1][0],
                      W2 + (size_t)(k0 + 32) * D + tn0, D);
        __builtin_amdgcn_s_wait_tensorcnt(1);
      } else {
        __builtin_amdgcn_s_wait_tensorcnt(0);
      }
    }
    __syncthreads();
#else
    stage_b_tile(&Bt[cur][0], W2 + (size_t)k0 * D + tn0, D, threadIdx.x);
    __syncthreads();
#endif
    v16bf a0 = load_a_frag(Arow0, K, k0, lane, 1.0f);
    v16bf a1 = load_a_frag(Arow1, K, k0, lane, 1.0f);
#pragma unroll
    for (int j = 0; j < 4; ++j) {
      v16bf bf = lds_b_frag(&Bt[cur][0], j * 16, lane);
      acc[0][j] = bfwmma(a0, bf, acc[0][j]);
      acc[1][j] = bfwmma(a1, bf, acc[1][j]);
    }
    __syncthreads();
  }

  const int n_base = tn0 + (lane & 15);
  const int m_base = tm0 + ((lane >> 4) << 3);
#pragma unroll
  for (int i = 0; i < 2; ++i)
#pragma unroll
    for (int r = 0; r < 8; ++r) {
      float g = grow[(size_t)(m_base + 16 * i + r) * E];
#pragma unroll
      for (int j = 0; j < 4; ++j) {
        size_t idx = (size_t)(m_base + 16 * i + r) * D + n_base + 16 * j;
        X[idx] += acc[i][j][r] * g;
      }
    }
}

// ---------- router: logits -> top-2 softmax gate (dense [Tp,E]) ----------
__global__ __launch_bounds__(32) void k_router(const float* __restrict__ Xn,
                                               const float* __restrict__ Rw,
                                               float* __restrict__ gate) {
  int row = blockIdx.x;
  int lane = threadIdx.x;
  const float* x = Xn + (size_t)row * D;
  float lg[E];
#pragma unroll
  for (int e = 0; e < E; ++e) lg[e] = 0.f;
  for (int d = lane; d < D; d += 32) {
    float t = x[d];
    const float* rw = Rw + (size_t)d * E;
#pragma unroll
    for (int e = 0; e < E; ++e) lg[e] += t * rw[e];
  }
#pragma unroll
  for (int e = 0; e < E; ++e) {
#pragma unroll
    for (int m = 1; m <= 16; m <<= 1) lg[e] += __shfl_xor(lg[e], m, 32);
  }
  if (lane == 0) {
    int i1 = 0;
#pragma unroll
    for (int e = 1; e < E; ++e) if (lg[e] > lg[i1]) i1 = e;
    int i2 = (i1 == 0) ? 1 : 0;
#pragma unroll
    for (int e = 0; e < E; ++e) if (e != i1 && lg[e] > lg[i2]) i2 = e;
    float w2  = expf(lg[i2] - lg[i1]);
    float inv = 1.f / (1.f + w2);
#pragma unroll
    for (int e = 0; e < E; ++e) {
      float g = (e == i1) ? inv : ((e == i2) ? w2 * inv : 0.f);
      gate[(size_t)row * E + e] = g;
    }
  }
}

// ---------- flash attention with sink, one wave per (b, h, 16-query tile) ----------
__global__ __launch_bounds__(32) void k_attn(const float* __restrict__ Q,
                                             const float* __restrict__ Km,
                                             const float* __restrict__ Vm,
                                             const float* __restrict__ sink,
                                             float* __restrict__ O) {
  const int qt = blockIdx.x, h = blockIdx.y, b = blockIdx.z;
  const int lane = threadIdx.x;
  const int half = lane >> 4, ln = lane & 15;
  const float qscale = rsqrtf((float)HD);

  const int qrow0 = b * Sp + qt * 16;
  const float* Qp = Q + (size_t)qrow0 * D + h * HD;
  v16bf qa0 = load_a_frag(Qp, D, 0,  lane, qscale);
  v16bf qa1 = load_a_frag(Qp, D, 32, lane, qscale);

  const float sv = sink[h];
  float mrow[8], lrow[8];
  v8f acc0 = vzero8(), acc1 = vzero8(), acc2 = vzero8(), acc3 = vzero8();
#pragma unroll
  for (int r = 0; r < 8; ++r) { mrow[r] = sv; lrow[r] = 1.f; }  // sink preseeds softmax

  __shared__ float Pl[16][33];

  for (int kt = 0; kt < Sp; kt += 32) {
    const float* Kp = Km + (size_t)(b * Sp + kt) * D + h * HD;
    v8f s0 = vzero8(), s1 = vzero8();
    {
      v16bf kb = load_a_frag(Kp, D, 0, lane, 1.f);
      s0 = bfwmma(qa0, kb, s0);
      kb = load_a_frag(Kp, D, 32, lane, 1.f);
      s0 = bfwmma(qa1, kb, s0);
      const float* Kp1 = Kp + (size_t)16 * D;
      kb = load_a_frag(Kp1, D, 0, lane, 1.f);
      s1 = bfwmma(qa0, kb, s1);
      kb = load_a_frag(Kp1, D, 32, lane, 1.f);
      s1 = bfwmma(qa1, kb, s1);
    }
    if (kt + ln >= S) {
#pragma unroll
      for (int r = 0; r < 8; ++r) s0[r] = -1e30f;
    }
    if (kt + 16 + ln >= S) {
#pragma unroll
      for (int r = 0; r < 8; ++r) s1[r] = -1e30f;
    }
    float p0[8], p1[8];
#pragma unroll
    for (int r = 0; r < 8; ++r) {
      float t = fmaxf(s0[r], s1[r]);
      t = fmaxf(t, __shfl_xor(t, 1, 32));
      t = fmaxf(t, __shfl_xor(t, 2, 32));
      t = fmaxf(t, __shfl_xor(t, 4, 32));
      t = fmaxf(t, __shfl_xor(t, 8, 32));
      float mnew  = fmaxf(mrow[r], t);
      float alpha = expf(mrow[r] - mnew);
      mrow[r] = mnew;
      p0[r] = expf(s0[r] - mnew);
      p1[r] = expf(s1[r] - mnew);
      float rs = p0[r] + p1[r];
      rs += __shfl_xor(rs, 1, 32);
      rs += __shfl_xor(rs, 2, 32);
      rs += __shfl_xor(rs, 4, 32);
      rs += __shfl_xor(rs, 8, 32);
      lrow[r] = lrow[r] * alpha + rs;
      acc0[r] *= alpha; acc1[r] *= alpha; acc2[r] *= alpha; acc3[r] *= alpha;
    }
    __syncthreads();
#pragma unroll
    for (int r = 0; r < 8; ++r) {
      Pl[r + 8 * half][ln]      = p0[r];
      Pl[r + 8 * half][16 + ln] = p1[r];
    }
    __syncthreads();
    v16bf pa;
#pragma unroll
    for (int e2 = 0; e2 < 16; ++e2) {
      int kk = (e2 & 7) + 8 * (((e2 >> 3) << 1) + half);
      pa[e2] = f2bf(Pl[ln][kk]);
    }
    const float* Vp = Vm + (size_t)(b * Sp + kt) * D + h * HD;
    acc0 = bfwmma(pa, load_b_strided(Vp, D, 0,  lane), acc0);
    acc1 = bfwmma(pa, load_b_strided(Vp, D, 16, lane), acc1);
    acc2 = bfwmma(pa, load_b_strided(Vp, D, 32, lane), acc2);
    acc3 = bfwmma(pa, load_b_strided(Vp, D, 48, lane), acc3);
  }
  float* Op = O + (size_t)qrow0 * D + h * HD;
#pragma unroll
  for (int r = 0; r < 8; ++r) {
    float inv = 1.f / lrow[r];
    int m = r + 8 * half;
    Op[(size_t)m * D + ln]      = acc0[r] * inv;
    Op[(size_t)m * D + 16 + ln] = acc1[r] * inv;
    Op[(size_t)m * D + 32 + ln] = acc2[r] * inv;
    Op[(size_t)m * D + 48 + ln] = acc3[r] * inv;
  }
}

// ---------- host orchestration ----------
static inline void launch_gemm(const float* A, int lda, const float* Bw, int ldb,
                               float* C, int ldc, int M, int N, int K,
                               const float* res, hipStream_t st) {
  dim3 g(N / BN, M / BM);
  k_gemm<<<g, 256, 0, st>>>(A, lda, Bw, ldb, C, ldc, K, res);
}

extern "C" void kernel_launch(void* const* d_in, const int* in_sizes, int n_in,
                              void* d_out, int out_size, void* d_ws, size_t ws_size,
                              hipStream_t stream) {
  (void)in_sizes; (void)n_in; (void)out_size; (void)ws_size;
  const float* patches = (const float*)d_in[0];
  const float* w_img   = (const float*)d_in[1];
  const float* cls     = (const float*)d_in[2];
  const float* storage = (const float*)d_in[3];
  const float* attnw   = (const float*)d_in[4];
  const float* ffnw    = (const float*)d_in[5];
  const float* wq      = (const float*)d_in[6];
  const float* wk      = (const float*)d_in[7];
  const float* wv      = (const float*)d_in[8];
  const float* wo      = (const float*)d_in[9];
  const float* sink    = (const float*)d_in[10];
  const float* router  = (const float*)d_in[11];
  const float* w1      = (const float*)d_in[12];
  const float* w2      = (const float*)d_in[13];
  const float* w3      = (const float*)d_in[14];
  const float* finw    = (const float*)d_in[15];

  float* ws = (float*)d_ws;
  const size_t NE = (size_t)Tp * D;
  float* x    = ws; ws += NE;
  float* hbuf = ws; ws += NE;
  float* xn   = ws; ws += NE;
  float* q    = ws; ws += NE;
  float* kb   = ws; ws += NE;
  float* vb   = ws; ws += NE;
  float* attn = ws; ws += NE;
  float* hh   = ws; ws += (size_t)Tp * F;
  float* gate = ws; ws += (size_t)Tp * E;   // ~76 MB total workspace

  // x = [cls | storage | patches@w_img], zero-padded to Sp rows per image
  k_zero<<<512, 256, 0, stream>>>(x, NE);
  k_fill<<<Bn * (1 + NS), 256, 0, stream>>>(x, cls, storage);
  for (int b = 0; b < Bn; ++b)
    launch_gemm(patches + (size_t)b * NPATCH * PDIM, PDIM, w_img, D,
                x + (size_t)(b * Sp + 1 + NS) * D, D, NPATCH, D, PDIM, nullptr, stream);

  for (int l = 0; l < L; ++l) {
    const float* wq_l = wq + (size_t)l * D * D;
    const float* wk_l = wk + (size_t)l * D * D;
    const float* wv_l = wv + (size_t)l * D * D;
    const float* wo_l = wo + (size_t)l * D * D;

    k_rmsnorm<<<Tp, 256, 0, stream>>>(x, attnw + (size_t)l * D, xn, D);
    launch_gemm(xn, D, wq_l, D, q,  D, Tp, D, D, nullptr, stream);
    launch_gemm(xn, D, wk_l, D, kb, D, Tp, D, D, nullptr, stream);
    launch_gemm(xn, D, wv_l, D, vb, D, Tp, D, D, nullptr, stream);
    k_rope<<<Tp, 256, 0, stream>>>(q, kb);

    dim3 ag(Sp / 16, H, Bn);
    k_attn<<<ag, 32, 0, stream>>>(q, kb, vb, sink + (size_t)l * H, attn);

    launch_gemm(attn, D, wo_l, D, hbuf, D, Tp, D, D, /*res=*/x, stream);

    k_rmsnorm<<<Tp, 256, 0, stream>>>(hbuf, ffnw + (size_t)l * D, xn, D);
    k_router<<<Tp, 32, 0, stream>>>(xn, router + (size_t)l * D * E, gate);
    k_copy<<<512, 256, 0, stream>>>(hbuf, x, NE);   // x := h, then accumulate MoE

    for (int e = 0; e < E; ++e) {
      const float* w1le = w1 + ((size_t)l * E + e) * (size_t)D * F;
      const float* w3le = w3 + ((size_t)l * E + e) * (size_t)D * F;
      const float* w2le = w2 + ((size_t)l * E + e) * (size_t)F * D;
      dim3 gu(F / BN, Tp / BM);
      k_moe_up<<<gu, 256, 0, stream>>>(xn, w1le, w3le, hh, D);
      dim3 gd(D / BN, Tp / BM);
      k_moe_down<<<gd, 256, 0, stream>>>(hh, w2le, x, gate + e, F);
    }
  }

  k_final<<<Bn * S, 256, 0, stream>>>(x, finw, (float*)d_out);
}
